// LuongConcatAttention_12412455485493
// MI455X (gfx1250) — matrix operand order repaired and checked by reading, not compiled, not run on hardware
//
#include <hip/hip_runtime.h>

typedef __attribute__((ext_vector_type(16))) __bf16 v16bf;
typedef __attribute__((ext_vector_type(8)))  float  v8f;

#define D_DIM 1024
#define S_DIM 2048
#define B_DIM 32
#define KC_N  32   // 1024 / 32 (K per wmma)
#define EC_N  64   // 1024 / 16 (N per wmma)

// round-to-nearest-even f32 -> bf16, packed pair into one dword
__device__ __forceinline__ unsigned f2bf_pk(float lo, float hi) {
  union { float f; unsigned u; } a, b;
  a.f = lo; b.f = hi;
  unsigned la = (a.u + 0x7FFFu + ((a.u >> 16) & 1u)) >> 16;
  unsigned lb = (b.u + 0x7FFFu + ((b.u >> 16) & 1u)) >> 16;
  return (la & 0xFFFFu) | (lb << 16);
}

// Branch-free tanh: prefer CDNA5 hardware v_tanh_f32; else 5-op exp2/rcp
// identity tanh(x) = 1 - 2/(exp2(2*log2e*x) + 1), which saturates to +/-1
// via exp2 -> {inf, 0} with no exec-mask divergence.
__device__ __forceinline__ float fast_tanh(float x) {
#if __has_builtin(__builtin_amdgcn_tanhf)
  return __builtin_amdgcn_tanhf(x);
#else
  float t = __builtin_amdgcn_exp2f(x * 2.885390081777927f /* 2*log2(e) */);
  return 1.0f - 2.0f * __builtin_amdgcn_rcpf(t + 1.0f);
#endif
}

// ---------------------------------------------------------------------------
// Kernel 1: pack We (= W_a[:,1024:]) into B-fragment bf16 layout.
// Fragment f = ec*32+kc. Per lane (col n = lane&15), halves i=0..15 hold
// K = kc*32 + i + (lane>=16 ? 16 : 0).  One thread -> one packed dword pair.
// ---------------------------------------------------------------------------
__global__ void pack_we_kernel(const float* __restrict__ Wa,
                               unsigned* __restrict__ wepk) {
  int idx  = blockIdx.x * blockDim.x + threadIdx.x;  // 0 .. 524287
  int p    = idx & 7;
  int lane = (idx >> 3) & 31;
  int f    = idx >> 8;            // 0..2047
  int kc   = f & 31;
  int ec   = f >> 5;
  int n    = lane & 15;
  int e    = ec * 16 + n;
  int k0   = kc * 32 + ((lane >> 4) << 4) + 2 * p;
  const float* src = Wa + (size_t)e * (2 * D_DIM) + D_DIM;  // We row e
  wepk[idx] = f2bf_pk(src[k0], src[k0 + 1]);
}

// ---------------------------------------------------------------------------
// Kernel 2: dec_proj[b,e] = sum_k dec[b,k] * W_a[e,k]   (tiny, f32 scalar)
// ---------------------------------------------------------------------------
__global__ void dec_proj_kernel(const float* __restrict__ dec,
                                const float* __restrict__ Wa,
                                float* __restrict__ dproj) {
  int b = blockIdx.y;
  int e = blockIdx.x * blockDim.x + threadIdx.x;
  const float* w = Wa + (size_t)e * (2 * D_DIM);
  const float* x = dec + b * D_DIM;
  float acc = 0.f;
  for (int k = 0; k < D_DIM; k += 4) {
    float4 wv = *(const float4*)(w + k);
    float4 xv = *(const float4*)(x + k);
    acc += wv.x * xv.x + wv.y * xv.y + wv.z * xv.z + wv.w * xv.w;
  }
  dproj[b * D_DIM + e] = acc;
}

// ---------------------------------------------------------------------------
// Kernel 3: fused scores.  128 threads = 4 waves; wave owns 16 s-rows.
// Stage enc rows once to LDS as bf16 A-fragments (per-lane private slots,
// no barriers needed), then 64x32 wmma_f32_16x16x32_bf16 with tanh*v epilogue.
// ---------------------------------------------------------------------------
__global__ void scores_kernel(const float* __restrict__ enc,
                              const unsigned* __restrict__ wepk,
                              const float* __restrict__ dproj,
                              const float* __restrict__ vvec,
                              float* __restrict__ scores) {
  extern __shared__ unsigned lds[];  // [4 waves][32 kc][32 lanes][8 dwords] = 128KB
  const int tid  = threadIdx.x;
  const int lane = tid & 31;
  const int wv   = tid >> 5;                 // wave in block, 0..3
  const int b    = blockIdx.y;
  const int s0   = (blockIdx.x * 4 + wv) * 16;
  const int m    = lane & 15;                // my A row within tile
  const int hi   = lane >> 4;                // lane half select

  // ---- stage my row into LDS as A-fragments (bf16) ----
  // A k order used here (contiguous, as in packed B): lane<16: K = 0..15; lane>=16: K = 16..31.
  const float* rowp = enc + ((size_t)b * S_DIM + (s0 + m)) * D_DIM;
  const int koff = hi << 4;  // 0 or 16
  for (int kc = 0; kc < KC_N; ++kc) {
    int kb = kc * 32 + koff;
    float4 f0 = *(const float4*)(rowp + kb);
    float4 f1 = *(const float4*)(rowp + kb + 4);
    float4 f2 = *(const float4*)(rowp + kb + 8);
    float4 f3 = *(const float4*)(rowp + kb + 12);
    unsigned* dst = lds + (((wv * KC_N + kc) * 32 + lane) << 3);
    dst[0] = f2bf_pk(f0.x, f0.y);
    dst[1] = f2bf_pk(f0.z, f0.w);
    dst[2] = f2bf_pk(f1.x, f1.y);
    dst[3] = f2bf_pk(f1.z, f1.w);
    dst[4] = f2bf_pk(f2.x, f2.y);
    dst[5] = f2bf_pk(f2.z, f2.w);
    dst[6] = f2bf_pk(f3.x, f3.y);
    dst[7] = f2bf_pk(f3.z, f3.w);
  }
  // Each lane reads back only its own slots -> same-thread dependence,
  // compiler inserts s_wait_dscnt; no workgroup barrier required.

  float part[8] = {0.f, 0.f, 0.f, 0.f, 0.f, 0.f, 0.f, 0.f};

  for (int ec = 0; ec < EC_N; ++ec) {
    v8f acc = {0.f, 0.f, 0.f, 0.f, 0.f, 0.f, 0.f, 0.f};
#pragma unroll 4
    for (int kc = 0; kc < KC_N; ++kc) {
      v16bf afr = *(const v16bf*)(lds  + (((wv * KC_N + kc) * 32 + lane) << 3));
      v16bf bfr = *(const v16bf*)(wepk + (((ec * KC_N + kc) * 32 + lane) << 3));
      acc = __builtin_amdgcn_wmma_f32_16x16x32_bf16(
          false, afr, false, bfr, (short)0, acc, false, false);
    }
    // D layout: lane holds column N = lane&15; rows M = j + 8*hi.
    int e = ec * 16 + m;
    float dp = dproj[b * D_DIM + e];
    float vv = vvec[e];
#pragma unroll
    for (int j = 0; j < 8; ++j)
      part[j] += vv * fast_tanh(dp + acc[j]);
  }

  // reduce across the 16 lanes sharing each row half
#pragma unroll
  for (int j = 0; j < 8; ++j) {
    float x = part[j];
    x += __shfl_xor(x, 1);
    x += __shfl_xor(x, 2);
    x += __shfl_xor(x, 4);
    x += __shfl_xor(x, 8);
    part[j] = x;
  }
  if (m == 0) {
    int row = s0 + (hi << 3);
#pragma unroll
    for (int j = 0; j < 8; ++j)
      scores[b * S_DIM + row + j] = part[j];
  }
}

// ---------------------------------------------------------------------------
// Kernel 4: in-place softmax over s=2048 per batch row (one block per b).
// ---------------------------------------------------------------------------
__global__ void softmax_kernel(float* __restrict__ alpha) {
  __shared__ float red[256];
  const int b   = blockIdx.x;
  const int tid = threadIdx.x;
  float* row = alpha + b * S_DIM;
  float vals[8];
  float mx = -3.4e38f;
#pragma unroll
  for (int i = 0; i < 8; ++i) {
    vals[i] = row[tid + i * 256];
    mx = fmaxf(mx, vals[i]);
  }
  red[tid] = mx;
  __syncthreads();
  for (int off = 128; off > 0; off >>= 1) {
    if (tid < off) red[tid] = fmaxf(red[tid], red[tid + off]);
    __syncthreads();
  }
  mx = red[0];
  __syncthreads();
  float sum = 0.f;
#pragma unroll
  for (int i = 0; i < 8; ++i) {
    vals[i] = __expf(vals[i] - mx);
    sum += vals[i];
  }
  red[tid] = sum;
  __syncthreads();
  for (int off = 128; off > 0; off >>= 1) {
    if (tid < off) red[tid] += red[tid + off];
    __syncthreads();
  }
  float inv = __builtin_amdgcn_rcpf(red[0]);
#pragma unroll
  for (int i = 0; i < 8; ++i)
    row[tid + i * 256] = vals[i] * inv;
}

// ---------------------------------------------------------------------------
extern "C" void kernel_launch(void* const* d_in, const int* in_sizes, int n_in,
                              void* d_out, int out_size, void* d_ws, size_t ws_size,
                              hipStream_t stream) {
  (void)in_sizes; (void)n_in; (void)out_size; (void)ws_size;
  const float* dec = (const float*)d_in[0];  // (32, 1024)
  const float* enc = (const float*)d_in[1];  // (32, 2048, 1024)
  const float* Wa  = (const float*)d_in[2];  // (1024, 2048)
  const float* v   = (const float*)d_in[3];  // (1024,)
  float* alpha = (float*)d_out;              // (32, 2048)

  unsigned char* ws = (unsigned char*)d_ws;
  unsigned* wepk = (unsigned*)ws;                          // 2 MB packed bf16 We
  float*    dproj = (float*)(ws + 2u * 1024u * 1024u);     // 128 KB dec_proj

  pack_we_kernel<<<524288 / 256, 256, 0, stream>>>(Wa, wepk);
  dec_proj_kernel<<<dim3(D_DIM / 256, B_DIM), 256, 0, stream>>>(dec, Wa, dproj);
  scores_kernel<<<dim3(S_DIM / 64, B_DIM), 128, 128 * 1024, stream>>>(
      enc, wepk, dproj, v, alpha);
  softmax_kernel<<<B_DIM, 256, 0, stream>>>(alpha);
}
